// SpatialAttention_63934883168807
// MI455X (gfx1250) — compile-verified, matrix operand order checked by reading, not run on hardware
//
#include <hip/hip_runtime.h>
#include <math.h>

// ---------------- problem constants ----------------
static constexpr int B    = 128;
static constexpr int CIN  = 273;   // input channels (GEMM K)
static constexpr int COUT = 270;   // output channels (GEMM M)
static constexpr int KF   = 32;    // fourier K
static constexpr int K2   = KF * KF;   // 1024
static constexpr int T    = 2048;

static constexpr int MPAD  = 272;  // COUT padded to 17*16
static constexpr int KPADG = 288;  // CIN padded to 9*32 (bf16 WMMA K-step 32)
static constexpr int NKT   = KPADG / 32;  // 9 K-steps
static constexpr int NT    = 64;   // T-tile per workgroup
static constexpr int KPADL = 296;  // LDS row stride in bf16 elems (592B: 16B-aligned, conflict-spread)

#define TWO_PI_F 6.283185307179586f

typedef float  v8f   __attribute__((ext_vector_type(8)));
typedef __bf16 v16bf __attribute__((ext_vector_type(16)));

struct alignas(16) U128 { unsigned int w[4]; };
struct U256 { U128 lo, hi; };   // 32 bytes == v16bf

__device__ __forceinline__ unsigned short f2bf(float f) {
  unsigned u = __builtin_bit_cast(unsigned, f);
  u = (u + 0x7FFFu + ((u >> 16) & 1u)) >> 16;   // round-to-nearest-even
  return (unsigned short)u;
}

__device__ __forceinline__ void sched_fence() {
#if __has_builtin(__builtin_amdgcn_sched_barrier)
  __builtin_amdgcn_sched_barrier(0);   // keep load batch above, WMMA chain below
#endif
}

// ---------------- K1: fourier basis ----------------
__global__ __launch_bounds__(256) void basis_kernel(
    const float* __restrict__ x, const float* __restrict__ y,
    float* __restrict__ cosb, float* __restrict__ sinb) {
  int idx = blockIdx.x * 256 + threadIdx.x;
  if (idx >= CIN * K2) return;
  int i = idx / K2, j = idx % K2;
  float kk = (float)(j % KF + 1);
  float ll = (float)(j / KF + 1);
  float phase = TWO_PI_F * (kk * x[i] + ll * y[i]);
  float s, c;
  __sincosf(phase, &s, &c);
  cosb[idx] = c;
  sinb[idx] = s;
}

// ---------------- K2: logits (one wave per output element) ----------------
__global__ __launch_bounds__(256) void logits_kernel(
    const float* __restrict__ zr, const float* __restrict__ zi,
    const float* __restrict__ cosb, const float* __restrict__ sinb,
    float* __restrict__ araw) {
  int g = blockIdx.x * 8 + (threadIdx.x >> 5);
  int lane = threadIdx.x & 31;
  if (g >= COUT * CIN) return;
  int o = g / CIN, i = g % CIN;
  const float* zrp = zr   + (size_t)o * K2;
  const float* zip = zi   + (size_t)o * K2;
  const float* cp  = cosb + (size_t)i * K2;
  const float* sp  = sinb + (size_t)i * K2;
  float acc = 0.f;
  for (int j = lane; j < K2; j += 32)
    acc = fmaf(zrp[j], cp[j], fmaf(zip[j], sp[j], acc));
#pragma unroll
  for (int off = 16; off; off >>= 1) acc += __shfl_xor(acc, off, 32);
  if (lane == 0) araw[g] = acc;
}

// ---------------- K3: mask + softmax + bf16 pack (zero-padded 272x288) -----
__global__ __launch_bounds__(288) void softmax_pack_kernel(
    const float* __restrict__ araw, const float* __restrict__ x,
    const float* __restrict__ y, const float* __restrict__ xd,
    const float* __restrict__ yd, unsigned short* __restrict__ apad) {
  __shared__ float red[16];
  const int o   = blockIdx.x;        // 0..271 (270,271 are pad rows)
  const int tid = threadIdx.x;       // 0..287
  const int wid = tid >> 5, lane = tid & 31;
  const bool valid = (o < COUT) && (tid < CIN);

  float logit = 0.f;
  if (valid) {
    logit = araw[o * CIN + tid];
    float dx = xd[0] - x[tid], dy = yd[0] - y[tid];
    if (dx * dx + dy * dy < 0.1f) logit = 0.f;   // zero, not -inf (as in ref)
  }
  // row max
  float v = valid ? logit : -__builtin_inff();
#pragma unroll
  for (int off = 16; off; off >>= 1) v = fmaxf(v, __shfl_xor(v, off, 32));
  if (lane == 0) red[wid] = v;
  __syncthreads();
  if (tid == 0) {
    float m = red[0];
    for (int w = 1; w < 9; ++w) m = fmaxf(m, red[w]);
    red[15] = m;
  }
  __syncthreads();
  float m = red[15];
  float e = valid ? __expf(logit - m) : 0.f;
  // row sum
  float s = e;
#pragma unroll
  for (int off = 16; off; off >>= 1) s += __shfl_xor(s, off, 32);
  __syncthreads();
  if (lane == 0) red[wid] = s;
  __syncthreads();
  if (tid == 0) {
    float t = 0.f;
    for (int w = 0; w < 9; ++w) t += red[w];
    red[14] = t;
  }
  __syncthreads();
  float prob = e / red[14];
  apad[(size_t)o * KPADG + tid] = valid ? f2bf(prob) : (unsigned short)0;
}

// ---------------- K4: fused convert + bf16 WMMA GEMM -----------------------
// out[b, o, t] = sum_i a[o,i] * X[b,i,t]
// Block = (t-tile, batch); stages X tile as bf16 transposed in LDS.
// Wave w: fixes N-tile (w & 3), strides M-tiles (mt = w>>2; mt += 2).
//   - 9 B fragments (72 VGPRs) loaded from LDS once per wave.
//   - Per M-tile: 18 batched global_load_b128 (sched-fenced so the waitcnt
//     pass emits incremental waits), then 9 chained WMMAs.
__global__ __launch_bounds__(256) void gemm_kernel(
    const float* __restrict__ X, const unsigned short* __restrict__ apad,
    float* __restrict__ out) {
  __shared__ __align__(16) unsigned short Xs[NT * KPADL];  // 37,888 B

  const int b  = blockIdx.y;
  const int t0 = blockIdx.x * NT;
  const int tid = threadIdx.x;

  // LDS fill: Xs[n][k] = bf16(X[b][k][t0+n]); K-pad rows zeroed.
  // float4 loads: 1024 elems per sweep, 18 sweeps, fully coalesced.
  for (int it = 0; it < (KPADG * NT) / (256 * 4); ++it) {
    int idx4 = it * 256 + tid;          // 0..4607
    int k = idx4 >> 4;                  // 0..287
    int n = (idx4 & 15) << 2;           // 0,4,...,60
    float4 v = make_float4(0.f, 0.f, 0.f, 0.f);
    if (k < CIN)
      v = *reinterpret_cast<const float4*>(X + ((size_t)b * CIN + k) * T + t0 + n);
    unsigned short* dst = Xs + n * KPADL + k;
    dst[0 * KPADL] = f2bf(v.x);
    dst[1 * KPADL] = f2bf(v.y);
    dst[2 * KPADL] = f2bf(v.z);
    dst[3 * KPADL] = f2bf(v.w);
  }
  __syncthreads();

  const int wave = tid >> 5;
  const int lane = tid & 31;
  const int l16  = lane & 15;
  const int ksel = (lane >> 4) << 3;           // 0 or 8 (16-bit A/B fragment layout)
  const int nt   = wave & 3;                   // this wave's N-tile
  const int mt0  = wave >> 2;                  // 0 or 1; mt strides by 2

  // Preload this wave's 9 B fragments from LDS (col fixed = nt*16 + l16).
  U256 Bf[NKT];
#pragma unroll
  for (int kt = 0; kt < NKT; ++kt) {
    const U128* bp =
        reinterpret_cast<const U128*>(Xs + (nt * 16 + l16) * KPADL + kt * 32 + ksel);
    Bf[kt].lo = bp[0];
    Bf[kt].hi = bp[2];
  }

  for (int mt = mt0; mt < MPAD / 16; mt += 2) {
    const int row = mt * 16 + l16;
    const unsigned short* arow = apad + (size_t)row * KPADG + ksel;

    // Batch-load all 9 A fragments for this M-tile (18 x b128 in flight).
    U256 Af[NKT];
#pragma unroll
    for (int kt = 0; kt < NKT; ++kt) {
      const U128* ap = reinterpret_cast<const U128*>(arow + kt * 32);
      Af[kt].lo = ap[0];
      Af[kt].hi = ap[2];
    }
    sched_fence();   // don't sink the loads into the WMMA chain

    v8f acc = {};
#pragma unroll
    for (int kt = 0; kt < NKT; ++kt) {
      acc = __builtin_amdgcn_wmma_f32_16x16x32_bf16(
          /*neg_a=*/false, __builtin_bit_cast(v16bf, Af[kt]),
          /*neg_b=*/false, __builtin_bit_cast(v16bf, Bf[kt]),
          /*c_mod=*/(short)0, acc, /*reuse_a=*/false, /*reuse_b=*/false);
    }

    // Store C tile: VGPR r -> row mt*16 + (lane>=16 ? 8 : 0) + r, col = lane&15.
    // Only the last M-tile (mt==16 -> rows 256..271) can exceed COUT=270.
    float* orow = out + ((size_t)b * COUT + mt * 16 + (lane >> 4) * 8) * T
                      + t0 + nt * 16 + l16;
    if (mt < (MPAD / 16) - 1) {
#pragma unroll
      for (int r = 0; r < 8; ++r) orow[(size_t)r * T] = acc[r];
    } else {
#pragma unroll
      for (int r = 0; r < 8; ++r) {
        int o = mt * 16 + (lane >> 4) * 8 + r;
        if (o < COUT) orow[(size_t)r * T] = acc[r];
      }
    }
  }
}

// ---------------- launch ----------------
extern "C" void kernel_launch(void* const* d_in, const int* in_sizes, int n_in,
                              void* d_out, int out_size, void* d_ws, size_t ws_size,
                              hipStream_t stream) {
  const float* X  = (const float*)d_in[0];
  const float* zr = (const float*)d_in[1];
  const float* zi = (const float*)d_in[2];
  const float* x  = (const float*)d_in[3];
  const float* y  = (const float*)d_in[4];
  const float* xd = (const float*)d_in[5];
  const float* yd = (const float*)d_in[6];
  float* out = (float*)d_out;

  // workspace layout (all 16B aligned): cos | sin | raw logits | bf16 A (padded)
  float* cosb = (float*)d_ws;
  float* sinb = cosb + (size_t)CIN * K2;
  float* araw = sinb + (size_t)CIN * K2;
  unsigned short* apad = (unsigned short*)(araw + 73712); // 270*273 rounded up for alignment

  basis_kernel<<<(CIN * K2 + 255) / 256, 256, 0, stream>>>(x, y, cosb, sinb);
  logits_kernel<<<(COUT * CIN + 7) / 8, 256, 0, stream>>>(zr, zi, cosb, sinb, araw);
  softmax_pack_kernel<<<MPAD, 288, 0, stream>>>(araw, x, y, xd, yd, apad);
  gemm_kernel<<<dim3(T / NT, B), 256, 0, stream>>>(X, apad, out);
}
